// GNNVotingModel_41266045780998
// MI455X (gfx1250) — matrix operand order repaired and checked by reading, not compiled, not run on hardware
//
#include <hip/hip_runtime.h>

typedef __attribute__((ext_vector_type(16))) __bf16 v16bf;
typedef __attribute__((ext_vector_type(8)))  __bf16 v8bf;
typedef __attribute__((ext_vector_type(8)))  float  v8f;

#define HID 64
#define BN_EPS 1e-5f

// ---------------- degree / dinv ----------------
__global__ __launch_bounds__(256) void k_deg_init(float* deg, int n) {
  int i = blockIdx.x * 256 + threadIdx.x;
  if (i < n) deg[i] = 1.0f;               // self-loop contributes 1
}

__global__ __launch_bounds__(256) void k_deg_count(const long long* __restrict__ ei,
                                                   float* deg, long long E) {
  long long e = (long long)blockIdx.x * 256 + threadIdx.x;
  if (e < E) atomicAdd(&deg[ei[E + e]], 1.0f);   // dst row of edge_index
}

__global__ __launch_bounds__(256) void k_rsqrt(float* d, int n) {
  int i = blockIdx.x * 256 + threadIdx.x;
  if (i < n) d[i] = rsqrtf(d[i]);
}

// ---- WMMA GEMM + fused self-loop seed:
//   H[N x 64]   = X[N x K] @ W[K x 64]
//   AGG[N x 64] = H * dinv^2          (seed for edge aggregation)
// Block = 128 threads (4 waves). Each wave owns a 16-row x 64-col tile.
template<int K>
__global__ __launch_bounds__(128) void k_gemm_wmma(const float* __restrict__ X,
                                                   const float* __restrict__ W,
                                                   const float* __restrict__ dinv,
                                                   float* __restrict__ H,
                                                   float* __restrict__ AGG,
                                                   int nrows) {
  constexpr int LDK = K + 8;              // pad: row stride 2*(K+8) B, 16B-aligned,
                                          // not a multiple of 256B -> no bank conflicts
  __shared__ __bf16 Wt[HID * LDK];        // transposed weights: Wt[n*LDK + k]
  const int tid = threadIdx.x;
  for (int i = tid; i < K * HID; i += 128) {
    int k = i >> 6, n = i & 63;           // W is row-major [K x 64]
    Wt[n * LDK + k] = (__bf16)W[i];
  }
  __syncthreads();

  const int lane = tid & 31;
  const int mr   = lane & 15;             // row within tile / col within n-tile
  const int kh   = lane >> 4;             // half-wave selector in K
  const long row0 = (long)blockIdx.x * 64 + (tid >> 5) * 16;
  if (row0 >= nrows) return;              // wave-uniform: EXEC stays all-ones for WMMA

  v8f acc[4] = {};                        // 4 n-tiles of 16 cols each
  const float* xrow = X + (row0 + mr) * (long)K;

  #pragma unroll
  for (int kk = 0; kk < K; kk += 32) {
    // A fragment (16x32 bf16): lane (mr,kh) holds K = kk+kh*8+j and kk+16+kh*8+j
    const float4* xp = (const float4*)(xrow + kk + kh * 8);
    float4 x0 = xp[0], x1 = xp[1];        // K = kk+kh*8 .. +7
    float4 x2 = xp[4], x3 = xp[5];        // K = kk+16+kh*8 .. +7
    v16bf a;
    a[0]=(__bf16)x0.x; a[1]=(__bf16)x0.y; a[2]=(__bf16)x0.z; a[3]=(__bf16)x0.w;
    a[4]=(__bf16)x1.x; a[5]=(__bf16)x1.y; a[6]=(__bf16)x1.z; a[7]=(__bf16)x1.w;
    a[8]=(__bf16)x2.x; a[9]=(__bf16)x2.y; a[10]=(__bf16)x2.z; a[11]=(__bf16)x2.w;
    a[12]=(__bf16)x3.x; a[13]=(__bf16)x3.y; a[14]=(__bf16)x3.z; a[15]=(__bf16)x3.w;

    #pragma unroll
    for (int nt = 0; nt < 4; ++nt) {
      // B fragment (32x16 bf16): lane holds col n = nt*16+mr, K = kk+kh*8 .. +7 and kk+16+kh*8 .. +7
      // -> two 16-byte chunks in transposed LDS: two ds_load_b128
      const __bf16* wp = &Wt[(nt * 16 + mr) * LDK + kk + kh * 8];
      v8bf blo = *(const v8bf*)wp;
      v8bf bhi = *(const v8bf*)(wp + 16);
      v16bf b = __builtin_shufflevector(blo, bhi,
                  0,1,2,3,4,5,6,7,8,9,10,11,12,13,14,15);
      acc[nt] = __builtin_amdgcn_wmma_f32_16x16x32_bf16(
          false, a, false, b, (short)0, acc[nt], false, false);
    }
  }

  // C/D layout: VGPR r -> row kh*8+r, col = lane&15 within each 16-col n-tile
  #pragma unroll
  for (int r = 0; r < 8; ++r) {
    long row = row0 + kh * 8 + r;
    float di = dinv[row];
    float sc = di * di;
    float c0 = acc[0][r], c1 = acc[1][r], c2 = acc[2][r], c3 = acc[3][r];
    float* o  = H   + row * HID + mr;
    float* ag = AGG + row * HID + mr;
    o[0]  = c0;      o[16] = c1;      o[32] = c2;      o[48] = c3;
    ag[0] = c0 * sc; ag[16] = c1 * sc; ag[32] = c2 * sc; ag[48] = c3 * sc;
  }
}

// ---------------- message passing (64-wide) ----------------
// 16 threads per edge, 4 features each (float4 gather, 4 adjacent f32 atomics)
__global__ __launch_bounds__(256) void k_edge64(const long long* __restrict__ ei,
                                                const float* __restrict__ h,
                                                const float* __restrict__ dinv,
                                                float* agg, long long E) {
  long long t = (long long)blockIdx.x * 256 + threadIdx.x;
  long long e = t >> 4;
  if (e >= E) return;
  int c = (int)(t & 15) * 4;
  long long s = ei[e];
  long long d = ei[E + e];
  float coeff = dinv[s] * dinv[d];
  float4 hv = *(const float4*)(h + s * HID + c);
  float* ap = agg + d * HID + c;
  atomicAdd(ap + 0, hv.x * coeff);
  atomicAdd(ap + 1, hv.y * coeff);
  atomicAdd(ap + 2, hv.z * coeff);
  atomicAdd(ap + 3, hv.w * coeff);
}

__global__ __launch_bounds__(256) void k_bn_relu(float* agg,
                                                 const float* __restrict__ b,
                                                 const float* __restrict__ g,
                                                 const float* __restrict__ be,
                                                 const float* __restrict__ m,
                                                 const float* __restrict__ v, int n) {
  long long t = (long long)blockIdx.x * 256 + threadIdx.x;
  int i = (int)(t >> 6);
  if (i >= n) return;
  int f = (int)(t & 63);
  float z = (agg[t] + b[f] - m[f]) * rsqrtf(v[f] + BN_EPS) * g[f] + be[f];
  agg[t] = z > 0.0f ? z : 0.0f;
}

// ---------------- output head (64 -> 1), fused self-loop seed ----------------
__global__ __launch_bounds__(256) void k_gemv(const float* __restrict__ h,
                                              const float* __restrict__ W3,
                                              const float* __restrict__ dinv,
                                              float* h3, float* agg, int n) {
  int i = blockIdx.x * 256 + threadIdx.x;
  if (i >= n) return;
  const float4* hr = (const float4*)(h + (long)i * HID);
  float acc = 0.0f;
  #pragma unroll
  for (int q = 0; q < 16; ++q) {
    float4 hv = hr[q];
    acc += hv.x * W3[4*q] + hv.y * W3[4*q+1] + hv.z * W3[4*q+2] + hv.w * W3[4*q+3];
  }
  float di = dinv[i];
  h3[i] = acc;
  agg[i] = acc * di * di;
}

__global__ __launch_bounds__(256) void k_edge1(const long long* __restrict__ ei,
                                               const float* __restrict__ h3,
                                               const float* __restrict__ dinv,
                                               float* agg, long long E) {
  long long e = (long long)blockIdx.x * 256 + threadIdx.x;
  if (e >= E) return;
  long long s = ei[e];
  long long d = ei[E + e];
  atomicAdd(&agg[d], h3[s] * dinv[s] * dinv[d]);
}

__global__ __launch_bounds__(256) void k_sigmoid(const float* __restrict__ agg,
                                                 const float* __restrict__ b3,
                                                 float* out, int n) {
  int i = blockIdx.x * 256 + threadIdx.x;
  if (i >= n) return;
  float z = agg[i] + b3[0];
  out[i] = 1.0f / (1.0f + __expf(-z));
}

// ---------------- launcher ----------------
extern "C" void kernel_launch(void* const* d_in, const int* in_sizes, int n_in,
                              void* d_out, int out_size, void* d_ws, size_t ws_size,
                              hipStream_t stream) {
  const float*     x   = (const float*)d_in[0];
  const long long* ei  = (const long long*)d_in[1];
  const float* W1  = (const float*)d_in[2];
  const float* b1  = (const float*)d_in[3];
  const float* W2  = (const float*)d_in[4];
  const float* b2  = (const float*)d_in[5];
  const float* W3  = (const float*)d_in[6];
  const float* b3  = (const float*)d_in[7];
  const float* g1  = (const float*)d_in[8];
  const float* be1 = (const float*)d_in[9];
  const float* m1  = (const float*)d_in[10];
  const float* v1  = (const float*)d_in[11];
  const float* g2  = (const float*)d_in[12];
  const float* be2 = (const float*)d_in[13];
  const float* m2  = (const float*)d_in[14];
  const float* v2  = (const float*)d_in[15];
  float* out = (float*)d_out;

  const int       N = in_sizes[0] / 128;
  const long long E = (long long)in_sizes[1] / 2;

  char* ws = (char*)d_ws;
  float* dinv = (float*)ws;                           // N floats
  float* h3   = (float*)(ws + (1u << 19));            // N floats
  float* agg3 = (float*)(ws + (1u << 20));            // N floats
  float* bufA = (float*)(ws + 3u * (1u << 19));       // N*64 floats (agg)
  float* bufB = (float*)(ws + 3u * (1u << 19) + (size_t)N * HID * sizeof(float)); // h

  unsigned nb   = (unsigned)((N + 255) / 256);
  unsigned nfb  = (unsigned)(((long long)N * HID + 255) / 256);
  unsigned ebS  = (unsigned)((E + 255) / 256);
  unsigned eb16 = (unsigned)((E * 16 + 255) / 256);
  unsigned gb   = (unsigned)((N + 63) / 64);

  // degrees -> dinv
  k_deg_init <<<nb,  256, 0, stream>>>(dinv, N);
  k_deg_count<<<ebS, 256, 0, stream>>>(ei, dinv, E);
  k_rsqrt    <<<nb,  256, 0, stream>>>(dinv, N);

  // layer 1: 128 -> 64   (GEMM writes h=bufB, seeds agg=bufA with self term)
  k_gemm_wmma<128><<<gb, 128, 0, stream>>>(x, W1, dinv, bufB, bufA, N);
  k_edge64 <<<eb16, 256, 0, stream>>>(ei, bufB, dinv, bufA, E);
  k_bn_relu<<<nfb,  256, 0, stream>>>(bufA, b1, g1, be1, m1, v1, N);

  // layer 2: 64 -> 64   (in-place agg seed over own input rows: safe, see notes)
  k_gemm_wmma<64><<<gb, 128, 0, stream>>>(bufA, W2, dinv, bufB, bufA, N);
  k_edge64 <<<eb16, 256, 0, stream>>>(ei, bufB, dinv, bufA, E);
  k_bn_relu<<<nfb,  256, 0, stream>>>(bufA, b2, g2, be2, m2, v2, N);

  // layer 3: 64 -> 1, sigmoid
  k_gemv   <<<nb,  256, 0, stream>>>(bufA, W3, dinv, h3, agg3, N);
  k_edge1  <<<ebS, 256, 0, stream>>>(ei, h3, dinv, agg3, E);
  k_sigmoid<<<nb,  256, 0, stream>>>(agg3, b3, out, N);
}
